// DSAMoE_88029649698822
// MI455X (gfx1250) — matrix operand order repaired and checked by reading, not compiled, not run on hardware
//
#include <hip/hip_runtime.h>
#include <hip/hip_bf16.h>
#include <math.h>

typedef __bf16 bf16;
typedef bf16 bf16x8  __attribute__((ext_vector_type(8)));
typedef bf16 bf16x16 __attribute__((ext_vector_type(16)));
typedef float f32x8  __attribute__((ext_vector_type(8)));

#define B_SZ   16
#define S_LEN  2048
#define D_DIM  1024
#define H_DIM  256
#define SD_DIM 14
#define NC     6
#define NG     3
#define NE     18
#define MT     64          // tokens per workgroup
#define XS_STRIDE 1032     // D + 8 pad: conflict-free ds_load_b128 A-frags
#define HS_STRIDE 264      // H + 8 pad
#define NTOK   (B_SZ * S_LEN)

union AFrag { bf16x16 v; bf16x8 h[2]; };

// ---------------- kernel 1: per-(b,d) mean over S ----------------
__global__ void colmean_kernel(const float* __restrict__ feat,
                               float* __restrict__ fmean) {
    int b = blockIdx.x >> 2;
    int d = ((blockIdx.x & 3) << 8) + threadIdx.x;
    const float* p = feat + (size_t)b * S_LEN * D_DIM + d;
    float s = 0.f;
    for (int i = 0; i < S_LEN; ++i) s += p[(size_t)i * D_DIM];
    fmean[b * D_DIM + d] = s * (1.0f / S_LEN);
}

// ------ kernel 2: f32 [E][K][N] -> bf16 [E][N][K] (tiled transpose) ------
__global__ void transpose_bf16_kernel(const float* __restrict__ src,
                                      bf16* __restrict__ dst, int K, int N) {
    __shared__ float tile[32][33];
    int e  = blockIdx.z;
    int kb = blockIdx.x << 5, nb = blockIdx.y << 5;
    int tx = threadIdx.x, ty = threadIdx.y;
    const float* sp = src + (size_t)e * K * N;
    bf16*        dp = dst + (size_t)e * K * N;
    #pragma unroll
    for (int i = ty; i < 32; i += 8)
        tile[i][tx] = sp[(size_t)(kb + i) * N + nb + tx];
    __syncthreads();
    #pragma unroll
    for (int i = ty; i < 32; i += 8)
        dp[(size_t)(nb + i) * K + kb + tx] = (bf16)tile[tx][i];
}

// ---------------- kernel 3: routers + bias term (1 block) ----------------
__global__ void router_kernel(const float* __restrict__ x_raw,
                              const float* __restrict__ fmean,
                              const float* __restrict__ cond_w,
                              const float* __restrict__ cond_b,
                              const float* __restrict__ stage_w,
                              const float* __restrict__ stage_b,
                              const float* __restrict__ up_b,
                              float* __restrict__ flatw,
                              float* __restrict__ biasterm,
                              float* __restrict__ out_cond,
                              float* __restrict__ out_stage,
                              float* __restrict__ out_loads,
                              float* __restrict__ out_lb) {
    __shared__ float xrm[B_SZ][SD_DIM];
    __shared__ float clog[B_SZ][NC];
    __shared__ float slog[B_SZ][NG];
    __shared__ float fw[B_SZ][NE];
    __shared__ float loads[NE];
    int t = threadIdx.x;

    if (t < B_SZ * SD_DIM) {                      // x_raw mean over S
        int b = t / SD_DIM, sd = t % SD_DIM;
        const float* p = x_raw + (size_t)b * S_LEN * SD_DIM + sd;
        float s = 0.f;
        for (int i = 0; i < S_LEN; ++i) s += p[i * SD_DIM];
        xrm[b][sd] = s * (1.0f / S_LEN);
    }
    if (t < B_SZ * NC) {                          // cond logits
        int b = t / NC, c = t % NC;
        float s = cond_b[c];
        const float* f = fmean + b * D_DIM;
        for (int d = 0; d < D_DIM; ++d) s += f[d] * cond_w[d * NC + c];
        clog[b][c] = s;
    }
    __syncthreads();
    if (t < B_SZ * NG) {                          // stage logits
        int b = t / NG, g = t % NG;
        float s = stage_b[g];
        for (int sd = 0; sd < SD_DIM; ++sd) s += xrm[b][sd] * stage_w[sd * NG + g];
        slog[b][g] = s;
    }
    __syncthreads();
    if (t < B_SZ) {                               // softmaxes + joint weights
        int b = t;
        float cw[NC], sw[NG];
        float mx = clog[b][0];
        for (int c = 1; c < NC; ++c) mx = fmaxf(mx, clog[b][c]);
        float sum = 0.f;
        for (int c = 0; c < NC; ++c) { cw[c] = expf(clog[b][c] - mx); sum += cw[c]; }
        for (int c = 0; c < NC; ++c) { cw[c] /= sum; out_cond[b * NC + c] = cw[c]; }
        mx = slog[b][0];
        for (int g = 1; g < NG; ++g) mx = fmaxf(mx, slog[b][g]);
        sum = 0.f;
        for (int g = 0; g < NG; ++g) { sw[g] = expf(slog[b][g] - mx); sum += sw[g]; }
        for (int g = 0; g < NG; ++g) { sw[g] /= sum; out_stage[b * NG + g] = sw[g]; }
        for (int c = 0; c < NC; ++c)
            for (int g = 0; g < NG; ++g) {
                float w = cw[c] * sw[g];
                fw[b][c * NG + g] = w;
                flatw[b * NE + c * NG + g] = w;
            }
    }
    __syncthreads();
    if (t < NE) {                                 // expert loads
        float s = 0.f;
        for (int b = 0; b < B_SZ; ++b) s += fw[b][t];
        s *= (1.0f / B_SZ);
        loads[t] = s;
        out_loads[t] = s;
    }
    __syncthreads();
    if (t == 0) {
        float s = 0.f;
        for (int e = 0; e < NE; ++e) s += loads[e] * loads[e];
        out_lb[0] = (float)NE * s * 0.01f;
    }
    for (int idx = t; idx < B_SZ * D_DIM; idx += 256) {   // bias term
        int b = idx / D_DIM, d = idx % D_DIM;
        float s = 0.f;
        for (int e = 0; e < NE; ++e) s += fw[b][e] * up_b[e * D_DIM + d];
        biasterm[idx] = s;
    }
}

// ---------------- kernel 4: fused MoE adapter (WMMA bf16) ----------------
__global__ __launch_bounds__(512)
void moe_main_kernel(const float* __restrict__ feat,
                     const bf16* __restrict__ dwT,   // [E][H][D] bf16
                     const bf16* __restrict__ uwT,   // [E][D][H] bf16
                     const float* __restrict__ down_b,
                     const float* __restrict__ flatw,
                     const float* __restrict__ biasterm,
                     float* __restrict__ out) {
    __shared__ bf16 Xs[MT][XS_STRIDE];
    __shared__ bf16 Hs[MT][HS_STRIDE];

    const int tid   = threadIdx.x;
    const int wave  = tid >> 5;          // 0..15
    const int lane  = tid & 31;
    const int lhalf = lane >> 4;         // 0/1
    const int l15   = lane & 15;
    const int a_kb  = lhalf * 16;        // A-frag K offset (contiguous K order)
    const int b_kb  = lhalf * 16;        // B-frag K offset (contiguous K order, same as A)

    const int token0 = blockIdx.x * MT;
    const int b      = token0 / S_LEN;

    // stage X tile f32 -> bf16 LDS
    const float* fbase = feat + (size_t)token0 * D_DIM;
    for (int i = tid; i < MT * D_DIM; i += 512) {
        int m = i >> 10, d = i & (D_DIM - 1);
        Xs[m][d] = (bf16)fbase[i];
    }
    __syncthreads();

    const f32x8 zero = {0.f, 0.f, 0.f, 0.f, 0.f, 0.f, 0.f, 0.f};
    f32x8 acc[4][4];                     // persistent: 4 m-tiles x 4 n-tiles
    #pragma unroll
    for (int mt = 0; mt < 4; ++mt)
        #pragma unroll
        for (int nt = 0; nt < 4; ++nt) acc[mt][nt] = zero;

    for (int e = 0; e < NE; ++e) {
        const float we = flatw[b * NE + e];
        // ---- down GEMM: this wave computes H columns [wave*16, wave*16+16)
        f32x8 hacc[4];
        #pragma unroll
        for (int mt = 0; mt < 4; ++mt) hacc[mt] = zero;
        {
            const bf16* brow =
                dwT + ((size_t)e * H_DIM + wave * 16 + l15) * D_DIM;
            for (int k0 = 0; k0 < D_DIM; k0 += 32) {
                bf16x16 bfrag = *(const bf16x16*)(brow + k0 + b_kb);
                #pragma unroll
                for (int mt = 0; mt < 4; ++mt) {
                    AFrag a;
                    const bf16* ar = &Xs[mt * 16 + l15][k0 + a_kb];
                    a.h[0] = *(const bf16x8*)ar;
                    a.h[1] = *(const bf16x8*)(ar + 8);
                    hacc[mt] = __builtin_amdgcn_wmma_f32_16x16x32_bf16(
                        false, a.v, false, bfrag, (short)0, hacc[mt],
                        false, false);
                }
            }
        }
        // bias + exact GELU + pre-scale by routing weight -> Hs
        {
            const int n  = wave * 16 + l15;
            const float bias = down_b[e * H_DIM + n];
            #pragma unroll
            for (int mt = 0; mt < 4; ++mt)
                #pragma unroll
                for (int j = 0; j < 8; ++j) {
                    float v = hacc[mt][j] + bias;
                    float g = 0.5f * v * (1.0f + erff(v * 0.70710678f));
                    Hs[mt * 16 + j + lhalf * 8][n] = (bf16)(g * we);
                }
        }
        __syncthreads();
        // ---- up GEMM: this wave owns D columns [wave*64, wave*64+64)
        for (int k0 = 0; k0 < H_DIM; k0 += 32) {
            AFrag a[4];
            #pragma unroll
            for (int mt = 0; mt < 4; ++mt) {
                const bf16* ar = &Hs[mt * 16 + l15][k0 + a_kb];
                a[mt].h[0] = *(const bf16x8*)ar;
                a[mt].h[1] = *(const bf16x8*)(ar + 8);
            }
            #pragma unroll
            for (int nt = 0; nt < 4; ++nt) {
                const int n = wave * 64 + nt * 16 + l15;
                bf16x16 bfrag = *(const bf16x16*)(
                    uwT + ((size_t)e * D_DIM + n) * H_DIM + k0 + b_kb);
                #pragma unroll
                for (int mt = 0; mt < 4; ++mt)
                    acc[mt][nt] = __builtin_amdgcn_wmma_f32_16x16x32_bf16(
                        false, a[mt].v, false, bfrag, (short)0, acc[mt][nt],
                        false, false);
            }
        }
        __syncthreads();   // Hs consumed before next expert overwrites it
    }

    // epilogue: out = acc + features(f32 exact) + bias_term
    const float* bias = biasterm + (size_t)b * D_DIM;
    #pragma unroll
    for (int mt = 0; mt < 4; ++mt)
        #pragma unroll
        for (int j = 0; j < 8; ++j) {
            const int m = mt * 16 + j + lhalf * 8;
            const size_t row = (size_t)(token0 + m) * D_DIM;
            #pragma unroll
            for (int nt = 0; nt < 4; ++nt) {
                const int n = wave * 64 + nt * 16 + l15;
                out[row + n] = acc[mt][nt][j] + feat[row + n] + bias[n];
            }
        }
}

// ---------------- host launcher ----------------
extern "C" void kernel_launch(void* const* d_in, const int* in_sizes, int n_in,
                              void* d_out, int out_size, void* d_ws, size_t ws_size,
                              hipStream_t stream) {
    const float* features = (const float*)d_in[0];
    const float* x_raw    = (const float*)d_in[1];
    const float* down_w   = (const float*)d_in[2];
    const float* down_b   = (const float*)d_in[3];
    const float* up_w     = (const float*)d_in[4];
    const float* up_b     = (const float*)d_in[5];
    const float* cond_w   = (const float*)d_in[6];
    const float* cond_b   = (const float*)d_in[7];
    const float* stage_w  = (const float*)d_in[8];
    const float* stage_b  = (const float*)d_in[9];

    float* out        = (float*)d_out;
    float* out_cond   = out + (size_t)NTOK * D_DIM;
    float* out_stage  = out_cond + B_SZ * NC;
    float* out_loads  = out_stage + B_SZ * NG;
    float* out_lb     = out_loads + NE;

    char*  ws       = (char*)d_ws;
    float* fmean    = (float*)ws;                              // 64 KB
    float* flatw    = (float*)(ws + 65536);                    // 1.2 KB (pad 2K)
    float* biasterm = (float*)(ws + 65536 + 2048);             // 64 KB
    bf16*  dwT      = (bf16*)(ws + 65536 + 2048 + 65536);      // 9 MB
    bf16*  uwT      = dwT + (size_t)NE * D_DIM * H_DIM;        // 9 MB

    colmean_kernel<<<B_SZ * 4, 256, 0, stream>>>(features, fmean);
    transpose_bf16_kernel<<<dim3(D_DIM / 32, H_DIM / 32, NE), dim3(32, 8), 0,
                            stream>>>(down_w, dwT, D_DIM, H_DIM);
    transpose_bf16_kernel<<<dim3(H_DIM / 32, D_DIM / 32, NE), dim3(32, 8), 0,
                            stream>>>(up_w, uwT, H_DIM, D_DIM);
    router_kernel<<<1, 256, 0, stream>>>(x_raw, fmean, cond_w, cond_b, stage_w,
                                         stage_b, up_b, flatw, biasterm,
                                         out_cond, out_stage, out_loads, out_lb);
    moe_main_kernel<<<NTOK / MT, 512, 0, stream>>>(features, dwT, uwT, down_b,
                                                   flatw, biasterm, out);
}
